// TokenGater_66958540144893
// MI455X (gfx1250) — compile-verified, matrix operand order checked by reading, not run on hardware
//
#include <hip/hip_runtime.h>
#include <math.h>

typedef __attribute__((ext_vector_type(16))) __bf16 v16bf;
typedef __attribute__((ext_vector_type(8)))  float  v8f;
typedef __attribute__((ext_vector_type(4)))  float  f4;

#define BATCH 32
#define NTOK  4096
#define CDIM  768
#define HDIM  256
#define KBLK  24                 // 768 / 32
#define NTILES ((BATCH * NTOK) / 16)   // 8192 row tiles

#define Y_ELEMS    ((size_t)BATCH * (NTOK + 1) * CDIM)   // 100,687,872
#define SCORES_OFF (Y_ELEMS + 4)
#define PROBS_OFF  (SCORES_OFF + (size_t)BATCH * NTOK)

// workspace float layout
#define WS_PSUM 0                 // [32]  sum of probs per batch
#define WS_PQ   32                // [1]   sum of p*(1-p) global
#define WS_INVD 40                // [32]  1/max(sum(1-p), eps) per batch
#define WS_BG   72                // [32*768] background accumulators
#define WS_F32_COUNT 32768        // 128 KB zeroed region
#define WS_W1P_BYTE_OFF ((size_t)WS_F32_COUNT * 4)

// ---------------------------------------------------------------------------
__global__ void zero_ws(float* __restrict__ ws) {
    int idx = blockIdx.x * 256 + threadIdx.x;
    if (idx < WS_F32_COUNT) ws[idx] = 0.0f;
}

// ---------------------------------------------------------------------------
// Repack W1 (f32 row-major [768,256]) into the bf16 B-operand register layout:
// flat half index = ((kb*16 + ct)*32 + lane)*16 + j, where for half j
// (v = j/2, p = j%2, hh = lane/16):
//   K = kb*32 + (v>=4 ? 16 : 0) + hh*8 + 2*(v&3) + p,  N = ct*16 + (lane&15)
// so each lane's 16 halves are 32 contiguous bytes at consume time.
__global__ void pack_w1(const float* __restrict__ W1, __bf16* __restrict__ W1p) {
    int idx = blockIdx.x * 256 + threadIdx.x;          // 24*16*32*16 = 196608
    if (idx >= KBLK * 16 * 32 * 16) return;
    int j    = idx & 15;
    int lane = (idx >> 4) & 31;
    int ct   = (idx >> 9) & 15;
    int kb   = idx >> 13;
    int v = j >> 1, p = j & 1;
    int hh = lane >> 4;
    int k = kb * 32 + ((v >= 4) ? 16 : 0) + hh * 8 + 2 * (v & 3) + p;
    int n = ct * 16 + (lane & 15);
    W1p[idx] = (__bf16)W1[(size_t)k * HDIM + n];
}

// ---------------------------------------------------------------------------
// Fused score MLP: h = gelu(x @ W1 + b1); score = h @ W2 + b2; p = sigmoid.
// One wave per 16-token tile; 16 column tiles * 24 K-steps = 384 WMMAs/wave.
__global__ void __launch_bounds__(256)
gater_scores_wmma(const float* __restrict__ x, const __bf16* __restrict__ W1p,
                  const float* __restrict__ b1, const float* __restrict__ W2,
                  const float* __restrict__ b2, float* __restrict__ out,
                  float* __restrict__ ws) {
    int lane = threadIdx.x & 31;
    int wave = threadIdx.x >> 5;
    int tile = blockIdx.x * 8 + wave;
    int token0 = tile * 16;
    int hh = lane >> 4;       // K-half selector for A/B operands
    int mrow = lane & 15;     // A row / C-D column within the 16-wide tile

    const float* xrow = x + (size_t)(token0 + mrow) * CDIM;
    const v16bf* Bp = (const v16bf*)W1p;

    v8f acc[16] = {};

    for (int kb = 0; kb < KBLK; ++kb) {
        // A operand: 16x32 bf16 tile, two contiguous 8-float chunks per lane
        const float* p0 = xrow + kb * 32 + hh * 8;
        v16bf a;
#pragma unroll
        for (int j = 0; j < 8; ++j) a[j] = (__bf16)p0[j];
#pragma unroll
        for (int j = 0; j < 8; ++j) a[8 + j] = (__bf16)p0[16 + j];

        if (kb + 1 < KBLK)   // hint next K-step's B tiles toward the caches
            __builtin_prefetch(&Bp[(size_t)((kb + 1) * 16) * 32 + lane], 0, 1);

#pragma unroll
        for (int ct = 0; ct < 16; ++ct) {
            v16bf b = Bp[(size_t)(kb * 16 + ct) * 32 + lane];
            acc[ct] = __builtin_amdgcn_wmma_f32_16x16x32_bf16(
                false, a, false, b, (short)0, acc[ct], false, false);
        }
    }

    // bias + exact GELU + dot with W2 (column n = ct*16 + mrow for this lane)
    float sacc[8];
#pragma unroll
    for (int r = 0; r < 8; ++r) sacc[r] = 0.0f;
#pragma unroll
    for (int ct = 0; ct < 16; ++ct) {
        int n = ct * 16 + mrow;
        float bias = b1[n];
        float w2v  = W2[n];
#pragma unroll
        for (int r = 0; r < 8; ++r) {
            float h = acc[ct][r] + bias;
            float g = 0.5f * h * (1.0f + erff(h * 0.70710678118654752f));
            sacc[r] += g * w2v;
        }
    }

    // reduce across the 16 lanes of each half (C/D: lanes 0-15 -> M=r, 16-31 -> M=8+r)
#pragma unroll
    for (int r = 0; r < 8; ++r) {
        float v = sacc[r];
        v += __shfl_xor(v, 1, 16);
        v += __shfl_xor(v, 2, 16);
        v += __shfl_xor(v, 4, 16);
        v += __shfl_xor(v, 8, 16);
        sacc[r] = v;
    }

    if (mrow == 0) {
        float bb = b2[0];
        int mbase = token0 + hh * 8;
        float sp = 0.0f, spq = 0.0f;
#pragma unroll
        for (int r = 0; r < 8; ++r) {
            float s = sacc[r] + bb;                 // TEMPERATURE == 1
            float p = 1.0f / (1.0f + expf(-s));
            out[SCORES_OFF + mbase + r] = s;
            out[PROBS_OFF  + mbase + r] = p;
            sp  += p;
            spq += p * (1.0f - p);
        }
        int bidx = token0 / NTOK;
        atomicAdd(&ws[WS_PSUM + bidx], sp);
        atomicAdd(&ws[WS_PQ], spq);
    }
}

// ---------------------------------------------------------------------------
__global__ void finalize_losses(const int* __restrict__ kptr,
                                float* __restrict__ out, float* __restrict__ ws) {
    int t = threadIdx.x;  // 32 threads, one per batch
    int k = kptr[0];
    k = k < 0 ? 0 : (k > NTOK ? NTOK : k);
    float target = (float)k / (float)NTOK;

    float s = ws[WS_PSUM + t];
    float d = s / (float)NTOK - target;
    float d2 = d * d;
#pragma unroll
    for (int m = 1; m < 32; m <<= 1) d2 += __shfl_xor(d2, m, 32);

    ws[WS_INVD + t] = 1.0f / fmaxf((float)NTOK - s, 1e-6f);

    if (t == 0) {
        float ls = d2 / (float)BATCH;
        float lb = ws[WS_PQ] / (float)(BATCH * NTOK);
        out[Y_ELEMS + 0] = 1.0f * ls + 0.01f * lb;  // aux
        out[Y_ELEMS + 1] = ls;                       // loss_sparsity
        out[Y_ELEMS + 2] = lb;                       // loss_binary
        out[Y_ELEMS + 3] = 0.0f;                     // loss_temp
    }
}

// ---------------------------------------------------------------------------
// Streaming pass: y[:, :N, :] = x * p ; bg_acc[b, c] += x * (1 - p).
// 192 threads, one float4 column group per thread (192*4 = 768 cols).
// probs staged into LDS via gfx1250 async global->LDS copy; x read (last use)
// and y write (write-once) are nontemporal to keep L2 for W1p/probs.
#define TOKPB 128
__global__ void __launch_bounds__(192)
scale_and_bg(const float* __restrict__ x, const float* __restrict__ probs,
             float* __restrict__ y, float* __restrict__ bg_acc) {
    __shared__ __align__(16) float sp[TOKPB];
    int b  = blockIdx.y;
    int n0 = blockIdx.x * TOKPB;
    int t  = threadIdx.x;

    // wave 0: 32 lanes x 16B async copy = 128 probs into LDS
    if (t < 32) {
        unsigned int lds_addr = (unsigned int)(uintptr_t)(&sp[t * 4]);
        unsigned long long gaddr =
            (unsigned long long)(uintptr_t)(probs + (size_t)b * NTOK + n0 + t * 4);
        asm volatile("global_load_async_to_lds_b128 %0, %1, off"
                     :: "v"(lds_addr), "v"(gaddr) : "memory");
        asm volatile("s_wait_asynccnt 0x0" ::: "memory");
    }
    __syncthreads();

    const f4* __restrict__ xr =
        (const f4*)(x + ((size_t)b * NTOK + n0) * CDIM) + t;
    f4* __restrict__ yr =
        (f4*)(y + ((size_t)b * (NTOK + 1) + n0) * CDIM) + t;

    f4 acc = (f4)0.0f;
    for (int i = 0; i < TOKPB; ++i) {
        float p = sp[i];
        float w = 1.0f - p;
        f4 v = __builtin_nontemporal_load(xr);
        __builtin_nontemporal_store(v * p, yr);
        acc += v * w;
        xr += CDIM / 4;
        yr += CDIM / 4;
    }

    float* dst = bg_acc + b * CDIM + t * 4;
    atomicAdd(dst + 0, acc[0]);
    atomicAdd(dst + 1, acc[1]);
    atomicAdd(dst + 2, acc[2]);
    atomicAdd(dst + 3, acc[3]);
}

// ---------------------------------------------------------------------------
__global__ void write_bg(const float* __restrict__ ws, float* __restrict__ y) {
    int idx = blockIdx.x * 256 + threadIdx.x;  // 32*768 = 24576
    if (idx >= BATCH * CDIM) return;
    int b = idx / CDIM, c = idx % CDIM;
    float v = ws[WS_BG + idx] * ws[WS_INVD + b];
    y[((size_t)b * (NTOK + 1) + NTOK) * CDIM + c] = v;
}

// ---------------------------------------------------------------------------
extern "C" void kernel_launch(void* const* d_in, const int* in_sizes, int n_in,
                              void* d_out, int out_size, void* d_ws, size_t ws_size,
                              hipStream_t stream) {
    const float* x  = (const float*)d_in[0];
    const float* W1 = (const float*)d_in[1];
    const float* b1 = (const float*)d_in[2];
    const float* W2 = (const float*)d_in[3];
    const float* b2 = (const float*)d_in[4];
    const int*   kp = (const int*)d_in[5];

    float* out = (float*)d_out;
    float* ws  = (float*)d_ws;
    __bf16* W1p = (__bf16*)((char*)d_ws + WS_W1P_BYTE_OFF);

    zero_ws<<<(WS_F32_COUNT + 255) / 256, 256, 0, stream>>>(ws);
    pack_w1<<<(KBLK * 16 * 32 * 16 + 255) / 256, 256, 0, stream>>>(W1, W1p);
    gater_scores_wmma<<<NTILES / 8, 256, 0, stream>>>(x, W1p, b1, W2, b2, out, ws);
    finalize_losses<<<1, 32, 0, stream>>>(kp, out, ws);
    scale_and_bg<<<dim3(NTOK / TOKPB, BATCH), 192, 0, stream>>>(
        x, out + PROBS_OFF, out, ws + WS_BG);
    write_bg<<<(BATCH * CDIM + 255) / 256, 256, 0, stream>>>(ws, out);
}
